// LSTM_24008867184951
// MI455X (gfx1250) — matrix operand order repaired and checked by reading, not compiled, not run on hardware
//
#include <hip/hip_runtime.h>
#include <hip/hip_bf16.h>
#include <math.h>

typedef __attribute__((ext_vector_type(16))) _Float16 v16h;
typedef __attribute__((ext_vector_type(8)))  _Float16 v8h;
typedef __attribute__((ext_vector_type(8)))  float    v8f;
typedef __attribute__((ext_vector_type(4)))  float    v4f;

#define SEQ_ 16384
#define B_   10
#define IN_  128
#define H_   10
#define G_   40          // 4*H
#define NPAD 48          // 40 gate cols padded to 3 tiles of 16
#define KP2  32          // padded K for layers 2/3 (K=10 -> 32)
#define SB_  (SEQ_ * B_) // 163840 rows, multiple of 16

// ---------------------------------------------------------------------------
// Weight prep: fp32 (40,K) -> f16 zero-padded (48,KPAD), once per launch.
// ---------------------------------------------------------------------------
__global__ void prep_weights(const float* __restrict__ W1,  // (40,128)
                             const float* __restrict__ W2,  // (40,10)
                             _Float16* __restrict__ Wh1,    // (48,128)
                             _Float16* __restrict__ Wh2) {  // (48,32)
  int i = blockIdx.x * blockDim.x + threadIdx.x;
  if (i < NPAD * IN_) {
    int n = i / IN_, k = i % IN_;
    Wh1[i] = (n < G_) ? (_Float16)W1[n * IN_ + k] : (_Float16)0.0f;
  } else if (i < NPAD * IN_ + NPAD * KP2) {
    int j = i - NPAD * IN_;
    int n = j / KP2, k = j % KP2;
    Wh2[j] = (n < G_ && k < H_) ? (_Float16)W2[n * H_ + k] : (_Float16)0.0f;
  }
}

// Zero the padded f16 activation buffer (pad cols must be 0 for WMMA).
__global__ void zero_fill(unsigned int* __restrict__ p, int ndw) {
  int i = blockIdx.x * blockDim.x + threadIdx.x;
  if (i < ndw) p[i] = 0u;
}

// ---------------------------------------------------------------------------
// Layer-1 projection: xg[M,40] = x[M,128] * Wh1^T + (bih+bhh)
// A from fp32 (K=128, all indices statically in range -> unguarded float4),
// B from padded f16 -> unguarded 32B loads. One wave = one 16x16 C tile.
// ---------------------------------------------------------------------------
__global__ void __launch_bounds__(256)
proj_wmma_l1(const float* __restrict__ A,    // (M,128) fp32
             const _Float16* __restrict__ W, // (48,128) f16 padded
             const float* __restrict__ bih, const float* __restrict__ bhh,
             float* __restrict__ xg,         // (M,40)
             int M) {
  int wave = (blockIdx.x * blockDim.x + threadIdx.x) >> 5;
  int lane = threadIdx.x & 31;
  int mt = wave / 3;
  int nt = wave % 3;
  if (mt * 16 >= M) return;  // wave-uniform

  int g = lane >> 4;
  int l = lane & 15;
  int n = nt * 16 + l;
  int row = mt * 16 + l;
  const float* ar = A + (size_t)row * IN_;
  const _Float16* wr = W + (size_t)n * IN_;

  v8f c = {};
#pragma unroll
  for (int k0 = 0; k0 < IN_; k0 += 32) {
    // A frag: 16 contiguous elements a[0..15] = k0+16g+{0..15}
    v4f f0 = *(const v4f*)(ar + k0 + 16 * g);
    v4f f1 = *(const v4f*)(ar + k0 + 16 * g + 4);
    v4f f2 = *(const v4f*)(ar + k0 + 16 * g + 8);
    v4f f3 = *(const v4f*)(ar + k0 + 16 * g + 12);
    v16h a;
#pragma unroll
    for (int i = 0; i < 4; ++i) {
      a[i]      = (_Float16)f0[i];
      a[4 + i]  = (_Float16)f1[i];
      a[8 + i]  = (_Float16)f2[i];
      a[12 + i] = (_Float16)f3[i];
    }
    // B frag: 16 contiguous halves at k0 + 16g
    v16h bm = *(const v16h*)(wr + k0 + 16 * g);
    c = __builtin_amdgcn_wmma_f32_16x16x32_f16(false, a, false, bm,
                                               (short)0, c, false, false);
  }

  if (n < G_) {
    float bsum = bih[n] + bhh[n];
#pragma unroll
    for (int r = 0; r < 8; ++r) {
      int m = 8 * g + r;
      xg[(size_t)(mt * 16 + m) * G_ + n] = c[r] + bsum;
    }
  }
}

// ---------------------------------------------------------------------------
// Layer-2/3 projection: xg[M,40] = yh[M,32pad] * Wh2^T + (bih+bhh)
// Fully padded f16 operands, single K-chunk, all loads unguarded.
// ---------------------------------------------------------------------------
__global__ void __launch_bounds__(256)
proj_wmma_l2(const _Float16* __restrict__ Ah, // (M,32) f16 padded
             const _Float16* __restrict__ W,  // (48,32) f16 padded
             const float* __restrict__ bih, const float* __restrict__ bhh,
             float* __restrict__ xg,          // (M,40)
             int M) {
  int wave = (blockIdx.x * blockDim.x + threadIdx.x) >> 5;
  int lane = threadIdx.x & 31;
  int mt = wave / 3;
  int nt = wave % 3;
  if (mt * 16 >= M) return;  // wave-uniform

  int g = lane >> 4;
  int l = lane & 15;
  int n = nt * 16 + l;
  int row = mt * 16 + l;

  v8h lo = *(const v8h*)(Ah + (size_t)row * KP2 + 16 * g);
  v8h hi = *(const v8h*)(Ah + (size_t)row * KP2 + 16 * g + 8);
  v16h a = __builtin_shufflevector(lo, hi, 0, 1, 2, 3, 4, 5, 6, 7,
                                   8, 9, 10, 11, 12, 13, 14, 15);
  v16h bm = *(const v16h*)(W + (size_t)n * KP2 + 16 * g);

  v8f c = {};
  c = __builtin_amdgcn_wmma_f32_16x16x32_f16(false, a, false, bm,
                                             (short)0, c, false, false);

  if (n < G_) {
    float bsum = bih[n] + bhh[n];
#pragma unroll
    for (int r = 0; r < 8; ++r) {
      int m = 8 * g + r;
      xg[(size_t)(mt * 16 + m) * G_ + n] = c[r] + bsum;
    }
  }
}

// ---------------------------------------------------------------------------
// Sequential LSTM scan: one workgroup, 1 barrier per step.
// Thread (b,k) owns c[b][k], computes its 4 gates from LDS h (double-buffered),
// writes relu(h) as f16 into padded yh (valid cols only; pads pre-zeroed).
// ---------------------------------------------------------------------------
__device__ inline float sigmoidf_(float x) { return 1.0f / (1.0f + __expf(-x)); }

__global__ void __launch_bounds__(128)
lstm_scan_kernel(const float* __restrict__ xg,   // (S,B,40)
                 const float* __restrict__ Whh,  // (40,10)
                 _Float16* __restrict__ yh,      // (S*B,32) padded, relu(h)
                 float* __restrict__ hstate,     // (100) in/out
                 float* __restrict__ cstate,     // (100) in/out
                 int S, int init_zero) {
  __shared__ float h_sh[2][B_ * H_];
  int tid = threadIdx.x;
  bool act = (tid < B_ * H_);
  int b = tid / H_;
  int k = tid % H_;

  float wreg[4][H_];
  float c = 0.0f;
  if (act) {
#pragma unroll
    for (int gi = 0; gi < 4; ++gi)
#pragma unroll
      for (int kk = 0; kk < H_; ++kk)
        wreg[gi][kk] = Whh[(size_t)(gi * H_ + k) * H_ + kk];
    float h0 = init_zero ? 0.0f : hstate[tid];
    c        = init_zero ? 0.0f : cstate[tid];
    h_sh[0][tid] = h0;
  }
  __syncthreads();

  for (int t = 0; t < S; ++t) {
    int cur = t & 1, nxt = cur ^ 1;
    if (act) {
      const float* xgt = xg + (size_t)t * (B_ * G_) + b * G_;
      if (t + 1 < S)
        __builtin_prefetch(xg + (size_t)(t + 1) * (B_ * G_) + b * G_ + k * 4, 0, 3);
      float gi = xgt[k];
      float gf = xgt[H_ + k];
      float gg = xgt[2 * H_ + k];
      float go = xgt[3 * H_ + k];
      const float* hrow = &h_sh[cur][b * H_];
#pragma unroll
      for (int kk = 0; kk < H_; ++kk) {
        float hv = hrow[kk];
        gi = fmaf(wreg[0][kk], hv, gi);
        gf = fmaf(wreg[1][kk], hv, gf);
        gg = fmaf(wreg[2][kk], hv, gg);
        go = fmaf(wreg[3][kk], hv, go);
      }
      c = sigmoidf_(gf) * c + sigmoidf_(gi) * tanhf(gg);
      float h = sigmoidf_(go) * tanhf(c);
      h_sh[nxt][tid] = h;
      yh[(size_t)(t * B_ + b) * KP2 + k] = (_Float16)fmaxf(h, 0.0f);
    }
    __syncthreads();
  }

  if (act) {
    hstate[tid] = h_sh[S & 1][tid];
    cstate[tid] = c;
  }
}

// ---------------------------------------------------------------------------
// Final Conv1d(H,1,1): out[s*B+b] = sum_h w[h]*y[s,b,h] + bias
// ---------------------------------------------------------------------------
__global__ void conv_reduce(const _Float16* __restrict__ yh,
                            const float* __restrict__ w,
                            const float* __restrict__ bias,
                            float* __restrict__ out, int n) {
  int i = blockIdx.x * blockDim.x + threadIdx.x;
  if (i < n) {
    const _Float16* yr = yh + (size_t)i * KP2;
    float acc = bias[0];
#pragma unroll
    for (int kk = 0; kk < H_; ++kk) acc = fmaf(w[kk], (float)yr[kk], acc);
    out[i] = acc;
  }
}

// ---------------------------------------------------------------------------
extern "C" void kernel_launch(void* const* d_in, const int* in_sizes, int n_in,
                              void* d_out, int out_size, void* d_ws, size_t ws_size,
                              hipStream_t stream) {
  const float* x     = (const float*)d_in[0];
  const float* Wih1  = (const float*)d_in[1];
  const float* Whh1  = (const float*)d_in[2];
  const float* bih1  = (const float*)d_in[3];
  const float* bhh1  = (const float*)d_in[4];
  const float* Wih2  = (const float*)d_in[5];
  const float* Whh2  = (const float*)d_in[6];
  const float* bih2  = (const float*)d_in[7];
  const float* bhh2  = (const float*)d_in[8];
  const float* convw = (const float*)d_in[9];
  const float* convb = (const float*)d_in[10];
  float* out = (float*)d_out;

  // Workspace layout (all offsets 256B-aligned)
  char* ws = (char*)d_ws;
  float*    xg  = (float*)ws;                                  // SB*40 f32
  char*     p1  = ws + (size_t)SB_ * G_ * sizeof(float);       // 26,214,400
  _Float16* yh  = (_Float16*)p1;                               // SB*32 f16
  char*     p2  = p1 + (size_t)SB_ * KP2 * sizeof(_Float16);   // +10,485,760
  _Float16* Wh1 = (_Float16*)p2;                               // 48*128 f16
  char*     p3  = p2 + (size_t)NPAD * IN_ * sizeof(_Float16);  // +12,288
  _Float16* Wh2 = (_Float16*)p3;                               // 48*32 f16
  char*     p4  = p3 + (size_t)NPAD * KP2 * sizeof(_Float16);  // +3,072
  float*    h   = (float*)p4;                                  // 128 f32
  float*    c   = h + 128;                                     // 128 f32

  // Prep: pad weights to f16, zero the padded activation buffer.
  {
    int nprep = NPAD * IN_ + NPAD * KP2;
    prep_weights<<<(nprep + 255) / 256, 256, 0, stream>>>(Wih1, Wih2, Wh1, Wh2);
    int ndw = SB_ * KP2 / 2;  // halves -> dwords
    zero_fill<<<(ndw + 255) / 256, 256, 0, stream>>>((unsigned int*)yh, ndw);
  }

  // Exact wave grid: (SB/16) M-tiles * 3 N-tiles, 8 waves per 256-thread block
  const dim3 pblk(256);
  const dim3 pgrd(((SB_ / 16) * 3) / 8);  // 3840 blocks, 30720 waves

  // Layer 1
  proj_wmma_l1<<<pgrd, pblk, 0, stream>>>(x, Wh1, bih1, bhh1, xg, SB_);
  lstm_scan_kernel<<<1, 128, 0, stream>>>(xg, Whh1, yh, h, c, SEQ_, 1);
  // Layer 2 (shared lstm2 weights)
  proj_wmma_l2<<<pgrd, pblk, 0, stream>>>(yh, Wh2, bih2, bhh2, xg, SB_);
  lstm_scan_kernel<<<1, 128, 0, stream>>>(xg, Whh2, yh, h, c, SEQ_, 0);
  // Layer 3 (shared lstm2 weights)
  proj_wmma_l2<<<pgrd, pblk, 0, stream>>>(yh, Wh2, bih2, bhh2, xg, SB_);
  lstm_scan_kernel<<<1, 128, 0, stream>>>(xg, Whh2, yh, h, c, SEQ_, 0);
  // Conv reduction
  conv_reduce<<<(SB_ + 255) / 256, 256, 0, stream>>>(yh, convw, convb, out, SB_);
}
